// VQEmbedding_54288386622116
// MI455X (gfx1250) — matrix operand order repaired and checked by reading, not compiled, not run on hardware
//
#include <hip/hip_runtime.h>
#include <hip/hip_bf16.h>
#include <math.h>

// ---------------------------------------------------------------------------
// VQ-VAE vector quantization for MI455X (gfx1250, wave32).
//   N = 32768 rows, D = 256, K = 1024 codes.
//   argmin_k ( ||e_k||^2 - 2 x.e_k ) via split-bf16 WMMA GEMM (3 wmma/k-step
//   recovers ~f32 accuracy). Code tiles double-buffered in LDS, filled by
//   CDNA5 async global->LDS loads (GLOBAL_LOAD_ASYNC_TO_LDS_B128, ASYNCcnt)
//   so no VGPR staging exists to spill; two interleaved WMMA accumulator
//   chains per wave cover XDL latency.
// ---------------------------------------------------------------------------

typedef __attribute__((ext_vector_type(16))) __bf16 v16bf;
typedef __attribute__((ext_vector_type(8)))  float  v8f;

#define VQ_D      256
#define VQ_K      1024
#define VQ_N      32768
#define VQ_ELEMS  8388608   // N * D
#define BROW      264       // padded LDS row stride (bf16): 528B = 4-bank shift
                            // per row -> conflict-free 16-lane ds_load_b128
#define TILE_H_EL 8448      // 32 rows * BROW elements (one array, one buffer)
#define BUF_EL    16896     // hi + lo regions per buffer (elements)
#define BUF_BYTES 33792     // per buffer
#define HILO_BYTES 16896    // byte offset of lo region within a buffer
#define ELO_BYTE_OFF 524288 // elo base relative to ehi base in workspace

static __device__ __forceinline__ v16bf ld16bf(const __bf16* p) {
    v16bf r;
    ((uint4*)&r)[0] = ((const uint4*)p)[0];
    ((uint4*)&r)[1] = ((const uint4*)p)[1];
    return r;
}

// Issue 8 async b128 global->LDS copies (64B hi + 64B lo per thread).
// GVS mode: gaddr = SGPR64 + VGPR32 + IOFFSET ; ldsaddr = VGPR32 + IOFFSET.
static __device__ __forceinline__ void async_tile_ld(
    unsigned ldsH, unsigned ldsL, unsigned gofH, unsigned gofL,
    unsigned long long sbase)
{
    asm volatile(
        "global_load_async_to_lds_b128 %0, %2, %4 offset:0\n"
        "global_load_async_to_lds_b128 %0, %2, %4 offset:16\n"
        "global_load_async_to_lds_b128 %0, %2, %4 offset:32\n"
        "global_load_async_to_lds_b128 %0, %2, %4 offset:48\n"
        "global_load_async_to_lds_b128 %1, %3, %4 offset:0\n"
        "global_load_async_to_lds_b128 %1, %3, %4 offset:16\n"
        "global_load_async_to_lds_b128 %1, %3, %4 offset:32\n"
        "global_load_async_to_lds_b128 %1, %3, %4 offset:48\n"
        :
        : "v"(ldsH), "v"(ldsL), "v"(gofH), "v"(gofL), "s"(sbase)
        : "memory");
}

// ---------------------------------------------------------------------------
// Phase 0: emb (f32) -> ehi/elo (bf16 split) + per-row ||e||^2.
// One wave per code row. 128 blocks x 256 threads (8 waves) = 1024 waves.
// ---------------------------------------------------------------------------
__global__ __launch_bounds__(256) void vq_prep_kernel(
    const float* __restrict__ emb, __bf16* __restrict__ ehi,
    __bf16* __restrict__ elo, float* __restrict__ e2)
{
    const int lane = threadIdx.x & 31;
    const int wave = threadIdx.x >> 5;
    const int row  = blockIdx.x * 8 + wave;     // 0..1023
    const float* er = emb + (size_t)row * VQ_D;
    float acc = 0.0f;
#pragma unroll
    for (int j = 0; j < 8; ++j) {
        const int k = j * 32 + lane;            // coalesced across the wave
        const float f = er[k];
        const __bf16 h = (__bf16)f;
        ehi[(size_t)row * VQ_D + k] = h;
        elo[(size_t)row * VQ_D + k] = (__bf16)(f - (float)h);
        acc += f * f;
    }
#pragma unroll
    for (int off = 16; off > 0; off >>= 1)
        acc += __shfl_xor(acc, off, 32);
    if (lane == 0) e2[row] = acc;
}

// ---------------------------------------------------------------------------
// Phase 1: argmin over codes.
// Block = 8 waves, each wave owns 16 rows of x (A fragments resident in
// VGPRs). 32 rounds of 32 codes; tiles double-buffered in dynamic LDS and
// filled by async global->LDS b128 loads issued one round ahead
// (s_wait_asynccnt 0x8 retires the current round, leaves the prefetch in
// flight during the WMMAs). Two independent WMMA chains per wave.
// 256 blocks x 256 threads = 2048 waves.
// ---------------------------------------------------------------------------
__global__ __launch_bounds__(256) void vq_argmin_kernel(
    const float* __restrict__ x, const __bf16* __restrict__ ehi,
    const __bf16* __restrict__ elo, const float* __restrict__ e2,
    int* __restrict__ indices)
{
    extern __shared__ __align__(16) __bf16 sDyn[];   // 2 * 33792 bytes

    const int t     = threadIdx.x;
    const int lane  = t & 31;
    const int wave  = t >> 5;
    const int mrow  = lane & 15;
    const int half  = lane >> 4;
    const int baseM = (blockIdx.x * 8 + wave) * 16;

    // ---- async staging geometry: thread -> (row lr, 64B chunk lc) ---------
    const int lr = t >> 3;                        // 0..31 row within tile
    const int lc = t & 7;                         // 0..7  (64B chunks)
    const unsigned gBaseOff = (unsigned)(lr * 512 + lc * 64);  // bytes in ehi
    const unsigned ldsBase  = (unsigned)(size_t)sDyn;          // LDS byte offset
    const unsigned tOff     = (unsigned)(lr * 528 + lc * 64);  // padded row
    const unsigned ldsH0 = ldsBase + tOff;
    const unsigned ldsL0 = ldsH0 + HILO_BYTES;
    const unsigned ldsH1 = ldsH0 + BUF_BYTES;
    const unsigned ldsL1 = ldsL0 + BUF_BYTES;
    const unsigned long long sbase = (unsigned long long)(size_t)ehi;

    // prologue: stage round 0 into buffer 0 (overlaps the A conversion below)
    async_tile_ld(ldsH0, ldsL0, gBaseOff, gBaseOff + ELO_BYTE_OFF, sbase);

    // ---- A fragments: 16x32 bf16 per k-step, hi+lo split, kept in VGPRs ----
    // A layout (matches B): lane half holds K={half*16+0..15} contiguous,
    // row M=lane%16.
    v16bf ahi[8], alo[8];
    {
        const float* xrow = x + (size_t)(baseM + mrow) * VQ_D;
#pragma unroll
        for (int ks = 0; ks < 8; ++ks) {
            const int k0 = ks * 32;
#pragma unroll
            for (int e = 0; e < 8; ++e) {
                const float f0 = xrow[k0 + half * 16 + e];
                const float f1 = xrow[k0 + half * 16 + 8 + e];
                const __bf16 h0 = (__bf16)f0, h1 = (__bf16)f1;
                ahi[ks][e]     = h0;  alo[ks][e]     = (__bf16)(f0 - (float)h0);
                ahi[ks][8 + e] = h1;  alo[ks][8 + e] = (__bf16)(f1 - (float)h1);
            }
        }
    }

    float bS[8];
    int   bI[8];
#pragma unroll
    for (int r = 0; r < 8; ++r) { bS[r] = 3.4e38f; bI[r] = 0; }

    for (int tt = 0; tt < 32; ++tt) {
        const int c32 = tt * 32;
        if (tt < 31) {              // prefetch round tt+1 into the other buffer
            const unsigned go = gBaseOff + (unsigned)(tt + 1) * 16384u;
            if ((tt & 1) == 0)
                async_tile_ld(ldsH1, ldsL1, go, go + ELO_BYTE_OFF, sbase);
            else
                async_tile_ld(ldsH0, ldsL0, go, go + ELO_BYTE_OFF, sbase);
            asm volatile("s_wait_asynccnt 0x8" ::: "memory");  // round tt done
        } else {
            asm volatile("s_wait_asynccnt 0x0" ::: "memory");
        }
        __syncthreads();            // round-tt tile visible to all waves

        // B layout (32x16 bf16): lane half holds K=half*16+e, column N=mrow
        const __bf16* bufB = sDyn + (tt & 1) * BUF_EL;
        const __bf16* p0h = bufB + mrow * BROW + half * 16;
        const __bf16* p0l = p0h + TILE_H_EL;
        const __bf16* p1h = p0h + 16 * BROW;
        const __bf16* p1l = p0l + 16 * BROW;
        v8f c0 = {}, c1 = {};
#pragma unroll
        for (int ks = 0; ks < 8; ++ks) {
            const v16bf b0h = ld16bf(p0h + ks * 32);
            const v16bf b1h = ld16bf(p1h + ks * 32);
            const v16bf b0l = ld16bf(p0l + ks * 32);
            const v16bf b1l = ld16bf(p1l + ks * 32);
            // two independent accumulator chains, interleaved
            c0 = __builtin_amdgcn_wmma_f32_16x16x32_bf16(false, ahi[ks], false, b0h,
                                                         (short)0, c0, false, false);
            c1 = __builtin_amdgcn_wmma_f32_16x16x32_bf16(false, ahi[ks], false, b1h,
                                                         (short)0, c1, false, false);
            c0 = __builtin_amdgcn_wmma_f32_16x16x32_bf16(false, alo[ks], false, b0h,
                                                         (short)0, c0, false, false);
            c1 = __builtin_amdgcn_wmma_f32_16x16x32_bf16(false, alo[ks], false, b1h,
                                                         (short)0, c1, false, false);
            c0 = __builtin_amdgcn_wmma_f32_16x16x32_bf16(false, ahi[ks], false, b0l,
                                                         (short)0, c0, false, false);
            c1 = __builtin_amdgcn_wmma_f32_16x16x32_bf16(false, ahi[ks], false, b1l,
                                                         (short)0, c1, false, false);
        }

        const int code0 = c32 + mrow, code1 = c32 + 16 + mrow;
        const float e20 = e2[code0], e21 = e2[code1];
#pragma unroll
        for (int r = 0; r < 8; ++r) {           // c[r] = dot(row 8*half+r, code)
            const float s0 = e20 - 2.0f * c0[r];
            const float s1 = e21 - 2.0f * c1[r];
            // merge the two chains first (code0 < code1: s0 wins ties)
            const float sw = (s1 < s0) ? s1 : s0;
            const int   iw = (s1 < s0) ? code1 : code0;
            if (sw < bS[r] || (sw == bS[r] && iw < bI[r])) { bS[r] = sw; bI[r] = iw; }
        }

        __syncthreads();            // all reads done before round tt+2 overwrites
    }

    // ---- argmin across the 16 lanes of each half --------------------------
#pragma unroll
    for (int off = 1; off < 16; off <<= 1) {
#pragma unroll
        for (int r = 0; r < 8; ++r) {
            const float os = __shfl_xor(bS[r], off, 32);
            const int   oi = __shfl_xor(bI[r], off, 32);
            if (os < bS[r] || (os == bS[r] && oi < bI[r])) { bS[r] = os; bI[r] = oi; }
        }
    }
    if (mrow == 0) {
#pragma unroll
        for (int r = 0; r < 8; ++r)
            indices[baseM + half * 8 + r] = bI[r];   // row m = 8*half + r
    }
}

// ---------------------------------------------------------------------------
// Phase 2: gather emb[idx] -> out, fused sum((x-q)^2) partials + histogram.
// 2048 blocks x 256 threads, 16 rows per block, float4 throughout.
// ---------------------------------------------------------------------------
__global__ __launch_bounds__(256) void vq_gather_kernel(
    const float* __restrict__ x, const float* __restrict__ emb,
    const int* __restrict__ indices, float* __restrict__ qout,
    float* __restrict__ partials, unsigned* __restrict__ counts)
{
    __shared__ int   sIdx[16];
    __shared__ float sRed[256];
    const int t        = threadIdx.x;
    const int blockRow = blockIdx.x * 16;
    if (t < 16) {
        const int idx = indices[blockRow + t];
        sIdx[t] = idx;
        atomicAdd(&counts[idx], 1u);            // integer atomics: deterministic
    }
    __syncthreads();

    float acc = 0.0f;
    const int col  = t & 63;                    // 64 float4 per row
    const int rsub = t >> 6;                    // 4 rows in flight
#pragma unroll
    for (int r4 = 0; r4 < 4; ++r4) {
        const int rloc = r4 * 4 + rsub;
        const int row  = blockRow + rloc;
        const int idx  = sIdx[rloc];
        const float4 xv = ((const float4*)(x   + (size_t)row * VQ_D))[col];
        const float4 ev = ((const float4*)(emb + (size_t)idx * VQ_D))[col];
        ((float4*)(qout + (size_t)row * VQ_D))[col] = ev;
        const float dx = xv.x - ev.x, dy = xv.y - ev.y;
        const float dz = xv.z - ev.z, dw = xv.w - ev.w;
        acc += dx * dx + dy * dy + dz * dz + dw * dw;
    }
    sRed[t] = acc;
    __syncthreads();
    for (int s = 128; s > 0; s >>= 1) {
        if (t < s) sRed[t] += sRed[t + s];
        __syncthreads();
    }
    if (t == 0) partials[blockIdx.x] = sRed[0]; // fixed slot: deterministic sum
}

// ---------------------------------------------------------------------------
// Phase 3: reduce 2048 partials -> loss; 1024 histogram bins -> perplexity.
// Single 256-thread block, fixed-order reductions.
// ---------------------------------------------------------------------------
__global__ __launch_bounds__(256) void vq_final_kernel(
    const float* __restrict__ partials, const unsigned* __restrict__ counts,
    float* __restrict__ out_scalars)
{
    __shared__ float sLo[256];
    __shared__ float sEn[256];
    const int t = threadIdx.x;
    float l = 0.0f;
#pragma unroll
    for (int i = 0; i < 8; ++i) l += partials[t * 8 + i];
    float ent = 0.0f;
#pragma unroll
    for (int i = 0; i < 4; ++i) {
        const float p = (float)counts[t * 4 + i] * (1.0f / (float)VQ_N);
        ent += p * logf(p + 1e-10f);
    }
    sLo[t] = l; sEn[t] = ent;
    __syncthreads();
    for (int s = 128; s > 0; s >>= 1) {
        if (t < s) { sLo[t] += sLo[t + s]; sEn[t] += sEn[t + s]; }
        __syncthreads();
    }
    if (t == 0) {
        // q_latent + 0.25*e_latent, both numerically mean((x-q)^2)
        out_scalars[0] = 1.25f * sLo[0] / (float)VQ_ELEMS;  // loss
        out_scalars[1] = expf(-sEn[0]);                     // perplexity
    }
}

// ---------------------------------------------------------------------------
extern "C" void kernel_launch(void* const* d_in, const int* in_sizes, int n_in,
                              void* d_out, int out_size, void* d_ws, size_t ws_size,
                              hipStream_t stream) {
    (void)in_sizes; (void)n_in; (void)out_size; (void)ws_size;
    const float* x   = (const float*)d_in[0];   // [64,512,256] f32
    const float* emb = (const float*)d_in[1];   // [1024,256]   f32
    float* qout = (float*)d_out;                // [8388608] quantized | loss | perplexity

    char* ws = (char*)d_ws;
    int*      indices  = (int*)      (ws);                       // 128 KB
    __bf16*   ehi      = (__bf16*)   (ws + 131072);              // 512 KB
    __bf16*   elo      = (__bf16*)   (ws + 131072 + 524288);     // 512 KB (ehi+524288!)
    float*    e2       = (float*)    (ws + 131072 + 1048576);    //   4 KB
    unsigned* counts   = (unsigned*) (ws + 131072 + 1052672);    //   4 KB
    float*    partials = (float*)    (ws + 131072 + 1056768);    //   8 KB

    hipMemsetAsync(counts, 0, VQ_K * sizeof(unsigned), stream);
    vq_prep_kernel  <<<128,  256, 0, stream>>>(emb, ehi, elo, e2);
    vq_argmin_kernel<<<256,  256, 2 * BUF_BYTES, stream>>>(x, ehi, elo, e2, indices);
    vq_gather_kernel<<<2048, 256, 0, stream>>>(x, emb, indices, qout, partials, counts);
    vq_final_kernel <<<1,    256, 0, stream>>>(partials, counts, qout + VQ_ELEMS);
}
